// NoFoDifformer_CL_13460427506362
// MI455X (gfx1250) — matrix operand order repaired and checked by reading, not compiled, not run on hardware
//
#include <hip/hip_runtime.h>
#include <hip/hip_bf16.h>
#include <math.h>

typedef __attribute__((ext_vector_type(16))) __bf16 v16bf;
typedef __attribute__((ext_vector_type(8)))  __bf16 v8bf;
typedef __attribute__((ext_vector_type(8)))  float  v8f;

#define NN_    8192
#define NFEAT_ 512
#define HD_    512
#define NC_    256
#define NHNC_  1024

#define ACT_NONE 0
#define ACT_RELU 1
#define ACT_GELU 2

// ---------------------------------------------------------------------------
// CDNA5 async memory->LDS copy (ASYNCcnt-tracked), 16 bytes per lane.
// ---------------------------------------------------------------------------
__device__ __forceinline__ void async_copy_b128(unsigned lds_off, const void* gaddr) {
  asm volatile("global_load_async_to_lds_b128 %0, %1, off"
               :: "v"(lds_off), "v"(gaddr) : "memory");
}
__device__ __forceinline__ void wait_async() {
  asm volatile("s_wait_asynccnt 0x0" ::: "memory");
}
__device__ __forceinline__ unsigned lds_off(const void* p) {
  return (unsigned)(size_t)p;   // generic LDS addr: low 32 bits = LDS offset
}

// ---------------------------------------------------------------------------
// Generic tiled WMMA GEMM:  C = act( scale*(A@B) + bias + resid )
// A logical [M,K] (stored [K,M] if TA), B logical [K,N] (stored [N,K] if TB).
// 256 threads = 8 waves; block tile 128x128, BK=32; wave strip 16x128
// (8x v_wmma_f32_16x16x32_bf16 per K-step, one A-frag reused 8x).
// Full tiles are staged with global_load_async_to_lds_b128.
// ---------------------------------------------------------------------------
template <bool TA, bool TB, int ACT, bool OUTBF>
__global__ __launch_bounds__(256) void gemm_bf16(
    const __bf16* __restrict__ A, const __bf16* __restrict__ B,
    const float* __restrict__ bias, const float* __restrict__ resid,
    void* __restrict__ Cout, int M, int N, int K, float outScale) {
  constexpr int BM = 128, BN = 128, BK = 32;
  __shared__ __align__(16) __bf16 As[BM][BK + 8];
  __shared__ __align__(16) __bf16 Bs[BK][BN + 8];

  const int tid  = threadIdx.x;
  const int wave = tid >> 5;
  const int lane = tid & 31;
  const int hl   = lane >> 4;      // half of wave (0/1)
  const int l15  = lane & 15;
  const int bm   = blockIdx.y * BM;
  const int bn   = blockIdx.x * BN;
  const int mBase = wave * 16;
  const unsigned asBase = lds_off(&As[0][0]);
  const unsigned bsBase = lds_off(&Bs[0][0]);
  const bool fullM = (bm + BM <= M);
  const bool fullN = (bn + BN <= N);

  v8f acc[8];
#pragma unroll
  for (int nc = 0; nc < 8; ++nc) acc[nc] = v8f{};

  for (int k0 = 0; k0 < K; k0 += BK) {
    const bool fullK = (k0 + BK <= K);
    bool usedAsync = false;

    // ---- stage A tile (BM x BK) ----
    if (!TA && fullM && fullK) {
      // contiguous rows: async 16B chunks (4 per row)
      for (int ch = tid; ch < (BM * BK) / 8; ch += 256) {
        int r = ch >> 2, c16 = ch & 3;
        async_copy_b128(asBase + r * (BK + 8) * 2 + c16 * 16,
                        (const void*)(A + (size_t)(bm + r) * K + k0 + c16 * 8));
      }
      usedAsync = true;
    } else if (TA && fullM && fullK) {
      // stored [K,M]: contiguous along M -> vector b128 loads, transpose in LDS
      for (int ch = tid; ch < (BM * BK) / 8; ch += 256) {
        int c = ch & (BK - 1), m8 = ch >> 5;
        v8bf v = *(const v8bf*)(A + (size_t)(k0 + c) * M + bm + m8 * 8);
#pragma unroll
        for (int j = 0; j < 8; ++j) As[m8 * 8 + j][c] = v[j];
      }
    } else {
      for (int i = tid; i < BM * BK; i += 256) {
        int r = i >> 5, c = i & 31;
        int gm = bm + r, gk = k0 + c;
        __bf16 v = (__bf16)0.f;
        if (gm < M && gk < K)
          v = TA ? A[(size_t)gk * M + gm] : A[(size_t)gm * K + gk];
        As[r][c] = v;
      }
    }
    // ---- stage B tile (BK x BN) ----
    if (!TB && fullN && fullK) {
      for (int ch = tid; ch < (BK * BN) / 8; ch += 256) {
        int r = ch >> 4, c16 = ch & 15;
        async_copy_b128(bsBase + r * (BN + 8) * 2 + c16 * 16,
                        (const void*)(B + (size_t)(k0 + r) * N + bn + c16 * 8));
      }
      usedAsync = true;
    } else {
      for (int i = tid; i < BK * BN; i += 256) {
        int r = i >> 7, c = i & 127;
        int gk = k0 + r, gn = bn + c;
        __bf16 v = (__bf16)0.f;
        if (gk < K && gn < N)
          v = TB ? B[(size_t)gn * K + gk] : B[(size_t)gk * N + gn];
        Bs[r][c] = v;
      }
    }
    if (usedAsync) wait_async();
    __syncthreads();

    // A fragment: 16x32, contiguous k order (k = 16*half + e), same as B fragment
    v16bf afrag;
    const int aRow = mBase + l15;
    const int kOff = hl * 16;
#pragma unroll
    for (int e = 0; e < 16; ++e)
      afrag[e] = As[aRow][e + kOff];

#pragma unroll
    for (int nc = 0; nc < 8; ++nc) {
      v16bf bfrag;
      const int bCol = nc * 16 + l15;
      const int kB = hl * 16;
#pragma unroll
      for (int e = 0; e < 16; ++e) bfrag[e] = Bs[kB + e][bCol];
      acc[nc] = __builtin_amdgcn_wmma_f32_16x16x32_bf16(
          false, afrag, false, bfrag, (short)0, acc[nc], false, false);
    }
    __syncthreads();
  }

  // epilogue: C layout VGPR r -> (M = r + 8*half, N = l15)
#pragma unroll
  for (int nc = 0; nc < 8; ++nc) {
#pragma unroll
    for (int r = 0; r < 8; ++r) {
      int row = bm + mBase + r + hl * 8;
      int col = bn + nc * 16 + l15;
      if (row < M && col < N) {
        float v = acc[nc][r] * outScale;
        if (bias)  v += bias[col];
        if (resid) v += resid[(size_t)row * N + col];
        if (ACT == ACT_RELU) v = fmaxf(v, 0.f);
        else if (ACT == ACT_GELU) v = 0.5f * v * (1.f + erff(v * 0.70710678118f));
        if (OUTBF) ((__bf16*)Cout)[(size_t)row * N + col] = (__bf16)v;
        else       ((float*) Cout)[(size_t)row * N + col] = v;
      }
    }
  }
}

// ---------------------------------------------------------------------------
// Online row-logsumexp of  scale * (A @ B^T),  A,B: [Nn,128] bf16.
// Never materializes the 8192x8192 matrix: WMMA tiles + running (max,sum).
// Block = 256 thr (8 waves), 128 rows per block, 64 columns per tile.
// Tiles staged via global_load_async_to_lds_b128.
// ---------------------------------------------------------------------------
__global__ __launch_bounds__(256) void lse_rows_kernel(
    const __bf16* __restrict__ A, const __bf16* __restrict__ B,
    float* __restrict__ lse, int Nn, float scale) {
  constexpr int KD = 128, BM = 128, BN = 64;
  __shared__ __align__(16) __bf16 As[BM][KD + 8];
  __shared__ __align__(16) __bf16 Bs[BN][KD + 8];

  const int tid = threadIdx.x;
  const int wave = tid >> 5, lane = tid & 31;
  const int hl = lane >> 4, l15 = lane & 15;
  const int bm = blockIdx.x * BM;
  const int mBase = wave * 16;
  const unsigned asBase = lds_off(&As[0][0]);
  const unsigned bsBase = lds_off(&Bs[0][0]);

  for (int ch = tid; ch < (BM * KD) / 8; ch += 256) {
    int r = ch >> 4, c16 = ch & 15;
    async_copy_b128(asBase + r * (KD + 8) * 2 + c16 * 16,
                    (const void*)(A + (size_t)(bm + r) * KD + c16 * 8));
  }
  wait_async();
  __syncthreads();

  // A fragments for all 4 K-steps, loaded once
  v16bf af[4];
  const int aRow = mBase + l15;
  const int kOff = hl * 16;
#pragma unroll
  for (int kc = 0; kc < 4; ++kc)
#pragma unroll
    for (int e = 0; e < 16; ++e)
      af[kc][e] = As[aRow][kc * 32 + e + kOff];

  float runMax[8], runSum[8];
#pragma unroll
  for (int r = 0; r < 8; ++r) { runMax[r] = -INFINITY; runSum[r] = 0.f; }

  const int ntiles = Nn / BN;
  for (int jt = 0; jt < ntiles; ++jt) {
    __syncthreads();
    for (int ch = tid; ch < (BN * KD) / 8; ch += 256) {
      int r = ch >> 4, c16 = ch & 15;
      async_copy_b128(bsBase + r * (KD + 8) * 2 + c16 * 16,
                      (const void*)(B + (size_t)(jt * BN + r) * KD + c16 * 8));
    }
    wait_async();
    __syncthreads();

    v8f acc[4] = {v8f{}, v8f{}, v8f{}, v8f{}};
#pragma unroll
    for (int nc = 0; nc < 4; ++nc) {
      const int bCol = nc * 16 + l15;
#pragma unroll
      for (int kc = 0; kc < 4; ++kc) {
        v16bf bf;
        const int kB = kc * 32 + hl * 16;
#pragma unroll
        for (int e = 0; e < 16; ++e) bf[e] = Bs[bCol][kB + e];
        acc[nc] = __builtin_amdgcn_wmma_f32_16x16x32_bf16(
            false, af[kc], false, bf, (short)0, acc[nc], false, false);
      }
    }
    // online logsumexp merge (reduce across the 16-lane column group)
#pragma unroll
    for (int r = 0; r < 8; ++r) {
      float v0 = acc[0][r] * scale, v1 = acc[1][r] * scale;
      float v2 = acc[2][r] * scale, v3 = acc[3][r] * scale;
      float tm = fmaxf(fmaxf(v0, v1), fmaxf(v2, v3));
      for (int s = 1; s < 16; s <<= 1) tm = fmaxf(tm, __shfl_xor(tm, s, 32));
      float ts = expf(v0 - tm) + expf(v1 - tm) + expf(v2 - tm) + expf(v3 - tm);
      for (int s = 1; s < 16; s <<= 1) ts += __shfl_xor(ts, s, 32);
      float nm = fmaxf(runMax[r], tm);
      runSum[r] = runSum[r] * expf(runMax[r] - nm) + ts * expf(tm - nm);
      runMax[r] = nm;
    }
  }
  if (l15 == 0) {
#pragma unroll
    for (int r = 0; r < 8; ++r)
      lse[bm + mBase + r + hl * 8] = runMax[r] + logf(runSum[r]);
  }
}

// ---------------------------------------------------------------------------
// Small kernels
// ---------------------------------------------------------------------------
__global__ void cvt_kernel(const float* __restrict__ in, __bf16* __restrict__ out,
                           size_t n) {
  for (size_t i = (size_t)blockIdx.x * blockDim.x + threadIdx.x; i < n;
       i += (size_t)gridDim.x * blockDim.x)
    out[i] = (__bf16)in[i];
}

__global__ void scale_cvt_kernel(const float* __restrict__ in,
                                 const float* __restrict__ rowScale,
                                 __bf16* __restrict__ out, size_t n, int cols) {
  for (size_t i = (size_t)blockIdx.x * blockDim.x + threadIdx.x; i < n;
       i += (size_t)gridDim.x * blockDim.x)
    out[i] = (__bf16)(in[i] * rowScale[i / cols]);
}

__global__ void add2_kernel(const float* a, const float* b, float* o, size_t n) {
  for (size_t i = (size_t)blockIdx.x * blockDim.x + threadIdx.x; i < n;
       i += (size_t)gridDim.x * blockDim.x)
    o[i] = a[i] + b[i];
}

__global__ void concat_kernel(const __bf16* a, const __bf16* b, __bf16* o, int Nn) {
  size_t n = (size_t)Nn * 2 * NC_;
  for (size_t i = (size_t)blockIdx.x * blockDim.x + threadIdx.x; i < n;
       i += (size_t)gridDim.x * blockDim.x) {
    size_t row = i >> 9; int col = (int)(i & 511);
    o[i] = (col < NC_) ? a[row * NC_ + col] : b[row * NC_ + (col - NC_)];
  }
}

__global__ void new_e_kernel(const float* __restrict__ e, const float* __restrict__ fd,
                             const float* __restrict__ fb, const float* __restrict__ rw,
                             const float* __restrict__ rb, const float* __restrict__ aw,
                             float* __restrict__ out, int Nn) {
  float freqs[16], c = 0.f;
#pragma unroll
  for (int f = 0; f < 16; ++f) {
    c += log1pf(expf(fd[f])) + 0.25f;                   // softplus + DMIN
    freqs[f] = 50.f * tanhf((c + fb[0]) / 50.f);        // OMEGA-band-limited
  }
  int i = blockIdx.x * blockDim.x + threadIdx.x;
  if (i >= Nn) return;
  float ev = e[i], pw = 1.f, accv = 0.f;
  for (int k = 0; k < 10; ++k) {
    pw *= ev;                                            // e^(k+1)
    float r = rw[k * 33] + rb[k];
#pragma unroll
    for (int f = 0; f < 16; ++f) {
      float ph = pw * freqs[f], s, cc;
      __sincosf(ph, &s, &cc);
      r += 0.25f * (rw[k * 33 + 1 + f] * s + rw[k * 33 + 17 + f] * cc);
    }
    accv += r * aw[k];
  }
  out[i] = accv;
}

__global__ __launch_bounds__(NC_) void layernorm_kernel(
    const float* __restrict__ in, const float* __restrict__ g,
    const float* __restrict__ b, __bf16* __restrict__ out, int C) {
  __shared__ float red[NC_];
  int row = blockIdx.x, c = threadIdx.x;
  float v = in[(size_t)row * C + c];
  red[c] = v; __syncthreads();
  for (int s = C >> 1; s > 0; s >>= 1) { if (c < s) red[c] += red[c + s]; __syncthreads(); }
  float mean = red[0] / C; __syncthreads();
  float d = v - mean;
  red[c] = d * d; __syncthreads();
  for (int s = C >> 1; s > 0; s >>= 1) { if (c < s) red[c] += red[c + s]; __syncthreads(); }
  float rstd = rsqrtf(red[0] / C + 1e-5f);
  out[(size_t)row * C + c] = (__bf16)(d * rstd * g[c] + b[c]);
}

__global__ __launch_bounds__(128) void rownorm_kernel(const float* __restrict__ in,
                                                      __bf16* __restrict__ out, int C) {
  __shared__ float red[128];
  int row = blockIdx.x, c = threadIdx.x;
  float v = in[(size_t)row * C + c];
  red[c] = v * v; __syncthreads();
  for (int s = C >> 1; s > 0; s >>= 1) { if (c < s) red[c] += red[c + s]; __syncthreads(); }
  float nrm = fmaxf(sqrtf(red[0]), 1e-12f);
  out[(size_t)row * C + c] = (__bf16)(v / nrm);
}

__global__ void diag_kernel(const __bf16* A, const __bf16* B, float* d,
                            int K, float scale, int Nn) {
  int i = blockIdx.x * blockDim.x + threadIdx.x;
  if (i >= Nn) return;
  float s = 0.f;
  for (int k = 0; k < K; ++k)
    s += (float)A[(size_t)i * K + k] * (float)B[(size_t)i * K + k];
  d[i] = s * scale;
}

__global__ __launch_bounds__(256) void loss_kernel(const float* li, const float* lj,
                                                   const float* dg, float* out, int Nn) {
  __shared__ float red[256];
  float s = 0.f;
  for (int i = threadIdx.x; i < Nn; i += 256)
    s += (li[i] - dg[i]) + (lj[i] - dg[i]);
  red[threadIdx.x] = s; __syncthreads();
  for (int st = 128; st > 0; st >>= 1) {
    if (threadIdx.x < st) red[threadIdx.x] += red[threadIdx.x + st];
    __syncthreads();
  }
  if (threadIdx.x == 0) out[0] = red[0] / (2.f * Nn);
}

// ---------------------------------------------------------------------------
extern "C" void kernel_launch(void* const* d_in, const int* in_sizes, int n_in,
                              void* d_out, int out_size, void* d_ws, size_t ws_size,
                              hipStream_t stream) {
  const float* e      = (const float*)d_in[0];
  const float* u      = (const float*)d_in[1];
  const float* x      = (const float*)d_in[2];
  const float* fe_w1  = (const float*)d_in[3];  const float* fe_b1 = (const float*)d_in[4];
  const float* fe_w2  = (const float*)d_in[5];  const float* fe_b2 = (const float*)d_in[6];
  const float* fd     = (const float*)d_in[7];  const float* fb    = (const float*)d_in[8];
  const float* read_w = (const float*)d_in[9];  const float* read_b= (const float*)d_in[10];
  const float* alphaw = (const float*)d_in[11];
  const float* ln1g   = (const float*)d_in[12]; const float* ln1b  = (const float*)d_in[13];
  const float* ln2g   = (const float*)d_in[14]; const float* ln2b  = (const float*)d_in[15];
  const float* wq = (const float*)d_in[16]; const float* bq = (const float*)d_in[17];
  const float* wk = (const float*)d_in[18]; const float* bk = (const float*)d_in[19];
  const float* wv = (const float*)d_in[20]; const float* bv = (const float*)d_in[21];
  const float* wo = (const float*)d_in[22]; const float* bo = (const float*)d_in[23];
  const float* f1w = (const float*)d_in[24]; const float* f1b = (const float*)d_in[25];
  const float* f2w = (const float*)d_in[26]; const float* f2b = (const float*)d_in[27];
  const float* pt1w = (const float*)d_in[28]; const float* pt1b = (const float*)d_in[29];
  const float* pt2w = (const float*)d_in[30]; const float* pt2b = (const float*)d_in[31];
  const float* pf1w = (const float*)d_in[32]; const float* pf1b = (const float*)d_in[33];
  const float* pf2w = (const float*)d_in[34]; const float* pf2b = (const float*)d_in[35];
  const float* mow = (const float*)d_in[36]; const float* mob = (const float*)d_in[37];

  char* ws = (char*)d_ws;
  size_t off = 0;
  auto alloc = [&](size_t bytes) -> void* {
    off = (off + 255) & ~(size_t)255;
    void* p = ws + off; off += bytes; return p;
  };
  const size_t NB = (size_t)NN_;

  __bf16* u_bf   = (__bf16*)alloc(NB * NB * 2);
  __bf16* x_bf   = (__bf16*)alloc(NB * NFEAT_ * 2);
  __bf16* w1b    = (__bf16*)alloc((size_t)NFEAT_ * HD_ * 2);
  __bf16* w2b    = (__bf16*)alloc((size_t)HD_ * NC_ * 2);
  __bf16* wqb    = (__bf16*)alloc((size_t)NC_ * NHNC_ * 2);
  __bf16* wkb    = (__bf16*)alloc((size_t)NC_ * NHNC_ * 2);
  __bf16* wvb    = (__bf16*)alloc((size_t)NC_ * NHNC_ * 2);
  __bf16* wob    = (__bf16*)alloc((size_t)NHNC_ * NC_ * 2);
  __bf16* f1wb   = (__bf16*)alloc((size_t)NC_ * NC_ * 2);
  __bf16* f2wb   = (__bf16*)alloc((size_t)NC_ * NC_ * 2);
  __bf16* pt1wb  = (__bf16*)alloc((size_t)NC_ * NC_ * 2);
  __bf16* pt2wb  = (__bf16*)alloc((size_t)NC_ * (NC_/2) * 2);
  __bf16* pf1wb  = (__bf16*)alloc((size_t)NC_ * NC_ * 2);
  __bf16* pf2wb  = (__bf16*)alloc((size_t)NC_ * (NC_/2) * 2);
  __bf16* mowb   = (__bf16*)alloc((size_t)2 * NC_ * NC_ * 2);
  __bf16* h1_bf  = (__bf16*)alloc(NB * HD_ * 2);
  float*  h_f    = (float*) alloc(NB * NC_ * 4);
  __bf16* h_bf   = (__bf16*)alloc(NB * NC_ * 2);
  float*  t0_f   = (float*) alloc(NB * NC_ * 4);
  __bf16* t0s_bf = (__bf16*)alloc(NB * NC_ * 2);
  float*  henc_f = (float*) alloc(NB * NC_ * 4);
  __bf16* hf_bf  = (__bf16*)alloc(NB * NC_ * 2);
  float*  newe_f = (float*) alloc(NB * 4);
  __bf16* mh_bf  = (__bf16*)alloc(NB * NC_ * 2);
  __bf16* q_bf   = (__bf16*)alloc(NB * NHNC_ * 2);
  __bf16* k_bf   = (__bf16*)alloc(NB * NHNC_ * 2);
  __bf16* v_bf   = (__bf16*)alloc(NB * NHNC_ * 2);
  __bf16* kv_bf  = (__bf16*)alloc((size_t)NHNC_ * NHNC_ * 2);
  __bf16* att_bf = (__bf16*)alloc(NB * NHNC_ * 2);
  float*  res_f  = (float*) alloc(NB * NC_ * 4);
  float*  mid_f  = (float*) alloc(NB * NC_ * 4);
  __bf16* fx_bf  = (__bf16*)alloc(NB * NC_ * 2);
  __bf16* g1_bf  = (__bf16*)alloc(NB * NC_ * 2);
  float*  ht_f   = (float*) alloc(NB * NC_ * 4);
  __bf16* ht_bf  = (__bf16*)alloc(NB * NC_ * 2);
  __bf16* p1_bf  = (__bf16*)alloc(NB * NC_ * 2);
  float*  zt_f   = (float*) alloc(NB * (NC_/2) * 4);
  float*  zf_f   = (float*) alloc(NB * (NC_/2) * 4);
  __bf16* ztn_bf = (__bf16*)alloc(NB * (NC_/2) * 2);
  __bf16* zfn_bf = (__bf16*)alloc(NB * (NC_/2) * 2);
  float*  lse_i  = (float*) alloc(NB * 4);
  float*  lse_j  = (float*) alloc(NB * 4);
  float*  dg_f   = (float*) alloc(NB * 4);
  __bf16* cat_bf = (__bf16*)alloc(NB * 2 * NC_ * 2);

  auto cvt = [&](const float* in, __bf16* out, size_t n) {
    unsigned blocks = (unsigned)((n + 1023) / 1024);
    cvt_kernel<<<blocks, 256, 0, stream>>>(in, out, n);
  };
  cvt(u, u_bf, NB * NB);
  cvt(x, x_bf, NB * NFEAT_);
  cvt(fe_w1, w1b, (size_t)NFEAT_ * HD_);
  cvt(fe_w2, w2b, (size_t)HD_ * NC_);
  cvt(wq, wqb, (size_t)NC_ * NHNC_);
  cvt(wk, wkb, (size_t)NC_ * NHNC_);
  cvt(wv, wvb, (size_t)NC_ * NHNC_);
  cvt(wo, wob, (size_t)NHNC_ * NC_);
  cvt(f1w, f1wb, (size_t)NC_ * NC_);
  cvt(f2w, f2wb, (size_t)NC_ * NC_);
  cvt(pt1w, pt1wb, (size_t)NC_ * NC_);
  cvt(pt2w, pt2wb, (size_t)NC_ * (NC_/2));
  cvt(pf1w, pf1wb, (size_t)NC_ * NC_);
  cvt(pf2w, pf2wb, (size_t)NC_ * (NC_/2));
  cvt(mow, mowb, (size_t)2 * NC_ * NC_);

  new_e_kernel<<<(NN_ + 255) / 256, 256, 0, stream>>>(e, fd, fb, read_w, read_b,
                                                      alphaw, newe_f, NN_);
  auto grid = [](int M, int N) { return dim3((N + 127) / 128, (M + 127) / 128); };

  // feat encoder
  gemm_bf16<false,false,ACT_RELU,true><<<grid(NN_, HD_), 256, 0, stream>>>(
      x_bf, w1b, fe_b1, nullptr, h1_bf, NN_, HD_, NFEAT_, 1.f);
  gemm_bf16<false,false,ACT_NONE,false><<<grid(NN_, NC_), 256, 0, stream>>>(
      h1_bf, w2b, fe_b2, nullptr, h_f, NN_, NC_, HD_, 1.f);
  cvt(h_f, h_bf, NB * NC_);

  // spectral propagation: henc = u @ (new_e * (u^T @ h));  hf == henc (ht==hf==h)
  gemm_bf16<true,false,ACT_NONE,false><<<grid(NN_, NC_), 256, 0, stream>>>(
      u_bf, h_bf, nullptr, nullptr, t0_f, NN_, NC_, NN_, 1.f);
  scale_cvt_kernel<<<(unsigned)((NB * NC_ + 1023) / 1024), 256, 0, stream>>>(
      t0_f, newe_f, t0s_bf, NB * NC_, NC_);
  gemm_bf16<false,false,ACT_NONE,false><<<grid(NN_, NC_), 256, 0, stream>>>(
      u_bf, t0s_bf, nullptr, nullptr, henc_f, NN_, NC_, NN_, 1.f);

  // linear attention
  layernorm_kernel<<<NN_, NC_, 0, stream>>>(h_f, ln1g, ln1b, mh_bf, NC_);
  gemm_bf16<false,false,ACT_NONE,true><<<grid(NN_, NHNC_), 256, 0, stream>>>(
      mh_bf, wqb, bq, nullptr, q_bf, NN_, NHNC_, NC_, 1.f);
  gemm_bf16<false,false,ACT_NONE,true><<<grid(NN_, NHNC_), 256, 0, stream>>>(
      mh_bf, wkb, bk, nullptr, k_bf, NN_, NHNC_, NC_, 1.f);
  gemm_bf16<false,false,ACT_NONE,true><<<grid(NN_, NHNC_), 256, 0, stream>>>(
      mh_bf, wvb, bv, nullptr, v_bf, NN_, NHNC_, NC_, 1.f);
  gemm_bf16<true,false,ACT_NONE,true><<<grid(NHNC_, NHNC_), 256, 0, stream>>>(
      k_bf, v_bf, nullptr, nullptr, kv_bf, NHNC_, NHNC_, NN_, 1.f);
  gemm_bf16<false,false,ACT_NONE,true><<<grid(NN_, NHNC_), 256, 0, stream>>>(
      q_bf, kv_bf, nullptr, nullptr, att_bf, NN_, NHNC_, NHNC_, 1.f);

  // mid = h + att@wo + bo + henc
  add2_kernel<<<(unsigned)((NB * NC_ + 1023) / 1024), 256, 0, stream>>>(
      h_f, henc_f, res_f, NB * NC_);
  gemm_bf16<false,false,ACT_NONE,false><<<grid(NN_, NC_), 256, 0, stream>>>(
      att_bf, wob, bo, res_f, mid_f, NN_, NC_, NHNC_, 1.f);

  // FFN with exact GELU
  layernorm_kernel<<<NN_, NC_, 0, stream>>>(mid_f, ln2g, ln2b, fx_bf, NC_);
  gemm_bf16<false,false,ACT_GELU,true><<<grid(NN_, NC_), 256, 0, stream>>>(
      fx_bf, f1wb, f1b, nullptr, g1_bf, NN_, NC_, NC_, 1.f);
  gemm_bf16<false,false,ACT_NONE,false><<<grid(NN_, NC_), 256, 0, stream>>>(
      g1_bf, f2wb, f2b, mid_f, ht_f, NN_, NC_, NC_, 1.f);
  cvt(ht_f, ht_bf, NB * NC_);
  cvt(henc_f, hf_bf, NB * NC_);   // hf == henc

  // projection heads
  gemm_bf16<false,false,ACT_RELU,true><<<grid(NN_, NC_), 256, 0, stream>>>(
      ht_bf, pt1wb, pt1b, nullptr, p1_bf, NN_, NC_, NC_, 1.f);
  gemm_bf16<false,false,ACT_NONE,false><<<grid(NN_, NC_/2), 256, 0, stream>>>(
      p1_bf, pt2wb, pt2b, nullptr, zt_f, NN_, NC_/2, NC_, 1.f);
  gemm_bf16<false,false,ACT_RELU,true><<<grid(NN_, NC_), 256, 0, stream>>>(
      hf_bf, pf1wb, pf1b, nullptr, p1_bf, NN_, NC_, NC_, 1.f);
  gemm_bf16<false,false,ACT_NONE,false><<<grid(NN_, NC_/2), 256, 0, stream>>>(
      p1_bf, pf2wb, pf2b, nullptr, zf_f, NN_, NC_/2, NC_, 1.f);
  rownorm_kernel<<<NN_, 128, 0, stream>>>(zt_f, ztn_bf, NC_/2);
  rownorm_kernel<<<NN_, 128, 0, stream>>>(zf_f, zfn_bf, NC_/2);

  // contrastive loss: streaming WMMA logsumexp over rows and columns of L
  lse_rows_kernel<<<NN_ / 128, 256, 0, stream>>>(ztn_bf, zfn_bf, lse_i, NN_, 2.0f);
  lse_rows_kernel<<<NN_ / 128, 256, 0, stream>>>(zfn_bf, ztn_bf, lse_j, NN_, 2.0f);
  diag_kernel<<<(NN_ + 255) / 256, 256, 0, stream>>>(ztn_bf, zfn_bf, dg_f,
                                                     NC_/2, 2.0f, NN_);

  // logits = [ht, hf] @ mo_w + mo_b  -> d_out[0 : N*NC]
  concat_kernel<<<(unsigned)((NB * 2 * NC_ + 1023) / 1024), 256, 0, stream>>>(
      ht_bf, hf_bf, cat_bf, NN_);
  gemm_bf16<false,false,ACT_NONE,false><<<grid(NN_, NC_), 256, 0, stream>>>(
      cat_bf, mowb, mob, nullptr, (float*)d_out, NN_, NC_, 2 * NC_, 1.f);

  // scalar loss -> d_out[N*NC]
  loss_kernel<<<1, 256, 0, stream>>>(lse_i, lse_j, dg_f,
                                     ((float*)d_out) + (size_t)NN_ * NC_, NN_);
  (void)in_sizes; (void)n_in; (void)out_size; (void)ws_size;
}
